// one_head_self_attention_88562225643969
// MI455X (gfx1250) — matrix operand order repaired and checked by reading, not compiled, not run on hardware
//
#include <hip/hip_runtime.h>

// Single-head attention: out = softmax((xWq^T+bq)(xWk^T+bk)^T / sqrt(D)) (xWv^T+bv)
// B=4, T=4096, D=128, fp32 in/out. Compute-bound (~36 GFLOP over ~16MB I/O):
// use bf16 V_WMMA_F32_16X16X32_BF16 (16384 FLOP/instr) with fp32 accumulation,
// flash-style online softmax, everything L2-resident (12MB bf16 working set).

#define B_ 4
#define T_ 4096
#define D_ 128
#define KB 64   // keys per flash block

using bf16 = __bf16;
typedef __attribute__((ext_vector_type(16))) __bf16 v16bf;
typedef __attribute__((ext_vector_type(8)))  __bf16 v8bf;
typedef __attribute__((ext_vector_type(8)))  float  v8f;
typedef __attribute__((ext_vector_type(4)))  float  v4f;

__device__ __forceinline__ v8f wmma_bf16(v16bf a, v16bf b, v8f c) {
    // (neg_a, A, neg_b, B, c_mod, C, reuse_a, reuse_b)
    return __builtin_amdgcn_wmma_f32_16x16x32_bf16(false, a, false, b, (short)0, c, false, false);
}

// ---------------------------------------------------------------------------
// Kernel 1: QKV projection (fp32 accumulate), writes bf16 q (scaled by
// log2(e)/sqrt(D)), k, v row-major. One block per 16 tokens, thread = out dim.
// ---------------------------------------------------------------------------
__global__ __launch_bounds__(128) void qkv_proj(
    const float* __restrict__ x,
    const float* __restrict__ Wq, const float* __restrict__ bq,
    const float* __restrict__ Wk, const float* __restrict__ bk,
    const float* __restrict__ Wv, const float* __restrict__ bv,
    bf16* __restrict__ qb, bf16* __restrict__ kb, bf16* __restrict__ vb)
{
    __shared__ float xs[16][D_];
    const int tid = threadIdx.x;
    const int blk = blockIdx.x;
    const int b   = blk / (T_ / 16);
    const int t0  = (blk % (T_ / 16)) * 16;
    const size_t base = (size_t)(b * T_ + t0) * D_;

#pragma unroll
    for (int r = 0; r < 16; ++r) xs[r][tid] = x[base + (size_t)r * D_ + tid];
    __syncthreads();

    const int o = tid;
    const float* __restrict__ wq = Wq + (size_t)o * D_;
    const float* __restrict__ wk = Wk + (size_t)o * D_;
    const float* __restrict__ wv = Wv + (size_t)o * D_;

    float aq[16], ak[16], av[16];
#pragma unroll
    for (int r = 0; r < 16; ++r) { aq[r] = bq[o]; ak[r] = bk[o]; av[r] = bv[o]; }

    for (int d = 0; d < D_; ++d) {
        const float wqv = wq[d], wkv = wk[d], wvv = wv[d];
#pragma unroll
        for (int r = 0; r < 16; ++r) {
            const float xv = xs[r][d];          // LDS broadcast (conflict-free)
            aq[r] = fmaf(xv, wqv, aq[r]);
            ak[r] = fmaf(xv, wkv, ak[r]);
            av[r] = fmaf(xv, wvv, av[r]);
        }
    }
    // Fold log2(e)/sqrt(128) into q so softmax can use native exp2.
    const float qscale = 1.4426950408889634f * 0.08838834764831845f;
#pragma unroll
    for (int r = 0; r < 16; ++r) {
        qb[base + (size_t)r * D_ + o] = (bf16)(aq[r] * qscale);
        kb[base + (size_t)r * D_ + o] = (bf16)(ak[r]);
        vb[base + (size_t)r * D_ + o] = (bf16)(av[r]);
    }
}

// ---------------------------------------------------------------------------
// Kernel 2: transpose v [b][t][d] -> vt [b][d][t] (bf16), 64x64 LDS tiles.
// ---------------------------------------------------------------------------
__global__ __launch_bounds__(256) void transpose_v(
    const bf16* __restrict__ vb, bf16* __restrict__ vt)
{
    __shared__ bf16 tile[64][64 + 2];
    const int b  = blockIdx.z;
    const int t0 = blockIdx.x * 64;
    const int d0 = blockIdx.y * 64;
    const int tx = threadIdx.x & 63;
    const int ty = threadIdx.x >> 6;   // 0..3
#pragma unroll
    for (int i = 0; i < 16; ++i) {
        const int t = ty + i * 4;
        tile[t][tx] = vb[(size_t)(b * T_ + t0 + t) * D_ + d0 + tx];
    }
    __syncthreads();
#pragma unroll
    for (int i = 0; i < 16; ++i) {
        const int d = ty + i * 4;
        vt[(size_t)(b * D_ + d0 + d) * T_ + t0 + tx] = tile[tx][d];
    }
}

// ---------------------------------------------------------------------------
// Kernel 3: flash attention. One wave32 per 16 queries; 4 waves/block.
// S^T = K*Q^T (so all global loads are contiguous), online base-2 softmax,
// O^T += V^T * P^T with V^T fragments from the pre-transposed vt buffer.
// No LDS, no barriers; 32 WMMA per 64-key block.
// ---------------------------------------------------------------------------
__global__ __launch_bounds__(128) void flash_attn(
    const bf16* __restrict__ qb, const bf16* __restrict__ kb,
    const bf16* __restrict__ vt, float* __restrict__ out)
{
    const int lane = threadIdx.x & 31;
    const int warp = threadIdx.x >> 5;
    const int n    = lane & 15;        // query column within 16-tile
    const int half = lane >> 4;        // 0: lanes 0-15, 1: lanes 16-31
    const bool up  = (half != 0);

    const int qt = blockIdx.x * 4 + warp;       // 0..1023
    const int b  = qt / (T_ / 16);
    const int t0 = (qt % (T_ / 16)) * 16;

    // Q as WMMA B-fragments: lane holds query n, 16 consecutive d per chunk.
    v16bf qfrag[4];
    {
        const bf16* qrow = qb + (size_t)(b * T_ + t0 + n) * D_;
#pragma unroll
        for (int ch = 0; ch < 4; ++ch)
            qfrag[ch] = *(const v16bf*)(qrow + ch * 32 + half * 16);
    }

    v8f oacc[8] = {};                  // O^T: 8 d-tiles x (16 d x 16 queries)
    float mrun = -3.0e38f, srun = 0.0f;

    const bf16* kbase = kb + (size_t)b * T_ * D_;
    const bf16* vbase = vt + (size_t)b * D_ * T_;

    for (int s0 = 0; s0 < T_; s0 += KB) {
        // ---- S^T = K * Q^T : 4 key tiles x 4 d-chunks = 16 WMMA ----
        v8f c4[4] = {};
#pragma unroll
        for (int t = 0; t < 4; ++t) {
            const bf16* krow = kbase + (size_t)(s0 + t * 16 + n) * D_;
#pragma unroll
            for (int ch = 0; ch < 4; ++ch) {
                // A operand in this code's contiguous k order (same as qfrag): lane half selects K 0-15 / 16-31
                v8bf lo = *(const v8bf*)(krow + ch * 32 + half * 16);
                v8bf hi = *(const v8bf*)(krow + ch * 32 + half * 16 + 8);
                v16bf a = __builtin_shufflevector(lo, hi,
                    0, 1, 2, 3, 4, 5, 6, 7, 8, 9, 10, 11, 12, 13, 14, 15);
                c4[t] = wmma_bf16(a, qfrag[ch], c4[t]);
            }
        }

        // ---- online softmax over this 64-key block (base 2) ----
        // Column n of S^T lives in lanes n (rows j) and n+16 (rows j+8).
        float mloc = c4[0][0];
#pragma unroll
        for (int t = 0; t < 4; ++t)
#pragma unroll
            for (int j = 0; j < 8; ++j) mloc = fmaxf(mloc, c4[t][j]);
        mloc = fmaxf(mloc, __shfl_xor(mloc, 16, 32));
        const float mnew  = fmaxf(mrun, mloc);
        const float alpha = __builtin_amdgcn_exp2f(mrun - mnew);
        mrun = mnew;

        float lsum = 0.0f;
#pragma unroll
        for (int t = 0; t < 4; ++t)
#pragma unroll
            for (int j = 0; j < 8; ++j) {
                const float p = __builtin_amdgcn_exp2f(c4[t][j] - mnew);
                c4[t][j] = p;                  // reuse S^T regs as P^T
                lsum += p;
            }
        lsum += __shfl_xor(lsum, 16, 32);
        srun = srun * alpha + lsum;

#pragma unroll
        for (int dt = 0; dt < 8; ++dt)
#pragma unroll
            for (int j = 0; j < 8; ++j) oacc[dt][j] *= alpha;

        // ---- O^T += V^T * P^T : repack P^T (C layout) into B layout, then
        //      8 d-tiles x 2 s-halves = 16 WMMA ----
#pragma unroll
        for (int f = 0; f < 2; ++f) {
            v16bf bp;
#pragma unroll
            for (int k = 0; k < 8; ++k) {
                // element k: s = f*32 + k (lower half) / f*32+16+k (upper half)
                const float swp = __shfl_xor(c4[2 * f + 1][k], 16, 32);
                bp[k] = (bf16)(up ? swp : c4[2 * f][k]);
            }
#pragma unroll
            for (int k = 0; k < 8; ++k) {
                // element 8+k: s = f*32+8+k (lower) / f*32+24+k (upper)
                const float swp = __shfl_xor(c4[2 * f][k], 16, 32);
                bp[8 + k] = (bf16)(up ? c4[2 * f + 1][k] : swp);
            }
#pragma unroll
            for (int dt = 0; dt < 8; ++dt) {
                const bf16* vrow = vbase + (size_t)(dt * 16 + n) * T_ + s0 + f * 32;
                v8bf lo = *(const v8bf*)(vrow + half * 16);
                v8bf hi = *(const v8bf*)(vrow + half * 16 + 8);
                v16bf a = __builtin_shufflevector(lo, hi,
                    0, 1, 2, 3, 4, 5, 6, 7, 8, 9, 10, 11, 12, 13, 14, 15);
                oacc[dt] = wmma_bf16(a, bp, oacc[dt]);
            }
        }
    }

    // ---- normalize and store: lane holds query n, d = dt*16 + half*8 + j ----
    const float inv = 1.0f / srun;
#pragma unroll
    for (int dt = 0; dt < 8; ++dt) {
        float* orow = out + (size_t)(b * T_ + t0 + n) * D_ + dt * 16 + half * 8;
        v4f lo, hi;
#pragma unroll
        for (int j = 0; j < 4; ++j) {
            lo[j] = oacc[dt][j] * inv;
            hi[j] = oacc[dt][4 + j] * inv;
        }
        *(v4f*)(orow)     = lo;
        *(v4f*)(orow + 4) = hi;
    }
}

// ---------------------------------------------------------------------------
extern "C" void kernel_launch(void* const* d_in, const int* in_sizes, int n_in,
                              void* d_out, int out_size, void* d_ws, size_t ws_size,
                              hipStream_t stream)
{
    (void)in_sizes; (void)n_in; (void)out_size; (void)ws_size;
    const float* x  = (const float*)d_in[0];
    const float* Wq = (const float*)d_in[1];
    const float* bq = (const float*)d_in[2];
    const float* Wk = (const float*)d_in[3];
    const float* bk = (const float*)d_in[4];
    const float* Wv = (const float*)d_in[5];
    const float* bv = (const float*)d_in[6];
    float* out = (float*)d_out;

    const size_t N = (size_t)B_ * T_ * D_;   // 2M elements
    bf16* qb = (bf16*)d_ws;                  // 4 MB
    bf16* kb = qb + N;                       // 4 MB
    bf16* vb = kb + N;                       // 4 MB
    bf16* vt = vb + N;                       // 4 MB  (total 16 MB workspace)

    qkv_proj<<<B_ * T_ / 16, 128, 0, stream>>>(x, Wq, bq, Wk, bk, Wv, bv, qb, kb, vb);
    transpose_v<<<dim3(T_ / 64, D_ / 64, B_), 256, 0, stream>>>(vb, vt);
    flash_attn<<<(B_ * T_ / 16) / 4, 128, 0, stream>>>(qb, kb, vt, out);
}
